// FusedOnlineAttention_28570122453866
// MI455X (gfx1250) — matrix operand order repaired and checked by reading, not compiled, not run on hardware
//
#include <hip/hip_runtime.h>
#include <math.h>

typedef __attribute__((ext_vector_type(16))) _Float16 v16h;
typedef __attribute__((ext_vector_type(8)))  _Float16 v8h;
typedef __attribute__((ext_vector_type(2)))  _Float16 v2h;
typedef __attribute__((ext_vector_type(8)))  float    v8f;
typedef __attribute__((ext_vector_type(4)))  unsigned int u32x4;
typedef __attribute__((ext_vector_type(8)))  int          i32x8;
typedef __attribute__((ext_vector_type(4)))  int          i32x4;

#define HEAD_DIM 128
#define QTILE    128   // q rows per block: 8 waves x 16
#define KTILE    64    // keys per iteration: four 16-wide WMMA N tiles
#define NTHREADS 256

#if __has_builtin(__builtin_amdgcn_tensor_load_to_lds) && __has_builtin(__builtin_amdgcn_s_wait_tensorcnt)
#define HAVE_TDM 1
#else
#define HAVE_TDM 0
#endif

__device__ __forceinline__ v8h cvt8(const float4 a, const float4 b) {
  v8h h;
  h[0] = (_Float16)a.x; h[1] = (_Float16)a.y; h[2] = (_Float16)a.z; h[3] = (_Float16)a.w;
  h[4] = (_Float16)b.x; h[5] = (_Float16)b.y; h[6] = (_Float16)b.z; h[7] = (_Float16)b.w;
  return h;
}

// LDS map (64 KB total):
//   [0,     32768)  K f32 staging (TDM destination) -- aliased by 8x2KB P bounce buffers
//   [32768, 49152)  K tile f16, row-major [64 k][128 d]
//   [49152, 65536)  V tile f16, transposed [128 d][64 k]
#define KSTAGE_OFF 0
#define PSH_OFF    0
#define KF16_OFF   32768
#define VT_OFF     49152

__launch_bounds__(NTHREADS, 1)
__global__ void fa_alibi_causal_kernel(const float* __restrict__ Q,
                                       const float* __restrict__ K,
                                       const float* __restrict__ V,
                                       const float* __restrict__ slopes,
                                       float* __restrict__ O,
                                       int B, int Sq, int Sk, int H)
{
  __shared__ __align__(128) unsigned char smem[65536];
  float*    Kstage = (float*)(smem + KSTAGE_OFF);
  _Float16* Kf16   = (_Float16*)(smem + KF16_OFF);
  _Float16* Vt     = (_Float16*)(smem + VT_OFF);

  const int tid   = threadIdx.x;
  const int wave  = tid >> 5;
  const int lane  = tid & 31;
  const int lhalf = lane >> 4;
  const int l16   = lane & 15;

  const int tilesPerBH = Sq / QTILE;
  const int tile = blockIdx.x % tilesPerBH;
  const int bh   = blockIdx.x / tilesPerBH;
  const int h    = bh % H;
  const int b    = bh / H;

  // log2(e) folded into scale and slope -> every exp is a bare v_exp_f32
  const float scale2  = 0.08838834764831845f * 1.4426950408889634f;
  const float slope2  = slopes[h] * 1.4426950408889634f;
  const float slope16 = slope2 * 16.0f;
  const int   q0      = tile * QTILE + wave * 16;

  // ---- Q rows once, straight into the 16-bit A-matrix lane layout ----
  v16h qfrag[4];
  {
    const int    qrow = q0 + l16;
    const float* qp   = Q + (((size_t)b * Sq + qrow) * H + h) * HEAD_DIM;
    #pragma unroll
    for (int c = 0; c < 4; ++c) {
      const int d0 = 32 * c + 8 * lhalf;
      const float4 f0 = *(const float4*)(qp + d0);
      const float4 f1 = *(const float4*)(qp + d0 + 4);
      const float4 f2 = *(const float4*)(qp + d0 + 16);
      const float4 f3 = *(const float4*)(qp + d0 + 20);
      v8h lo = cvt8(f0, f1);
      v8h hi = cvt8(f2, f3);
      qfrag[c] = __builtin_shufflevector(lo, hi, 0,1,2,3,4,5,6,7,8,9,10,11,12,13,14,15);
    }
  }

  const v8f zero = {0.f,0.f,0.f,0.f,0.f,0.f,0.f,0.f};
  v8f o[8];
  #pragma unroll
  for (int t = 0; t < 8; ++t) o[t] = zero;
  float mrow[8], lrow[8];
  #pragma unroll
  for (int r = 0; r < 8; ++r) { mrow[r] = -1e30f; lrow[r] = 0.f; }

  // all-ones f16 B fragment: rowsum(P) via WMMA instead of lane shuffles
  v16h onesf;
  #pragma unroll
  for (int j = 0; j < 16; ++j) onesf[j] = (_Float16)1.0f;

  const int kend = (Sk < (tile + 1) * QTILE) ? Sk : (tile + 1) * QTILE;

  for (int k0 = 0; k0 < kend; k0 += KTILE) {
    __syncthreads();   // previous iteration's LDS consumers are done

#if HAVE_TDM
    // ---- Tensor Data Mover: DMA the 64x128 f32 K tile into LDS staging ----
    if (wave == 0) {
      const unsigned long long ga =
          (unsigned long long)(uintptr_t)(K + (((size_t)b * Sk + k0) * H + h) * HEAD_DIM);
      const unsigned ldsw = (unsigned)(uintptr_t)(smem + KSTAGE_OFF);
      u32x4 g0;
      g0[0] = 1u;                                            // count=1, user mode
      g0[1] = ldsw;                                          // lds_addr
      g0[2] = (unsigned)ga;                                  // global_addr[31:0]
      g0[3] = (unsigned)((ga >> 32) & 0x01FFFFFFu)           // global_addr[56:32]
            | 0x80000000u;                                   // type=2 (image)
      i32x8 g1;
      g1[0] = (int)(2u << 16);                               // wg_mask=0, data_size=4B
      g1[1] = (int)(128u << 16);                             // tensor_dim0 = 128 (d)
      g1[2] = (int)(((unsigned)(Sk - k0) & 0xFFFFu) << 16);  // tensor_dim1 lo16 (rows left)
      g1[3] = (int)(128u << 16);                             // tile_dim0 = 128
      g1[4] = (int)64;                                       // tile_dim1 = 64, tile_dim2 = 0
      g1[5] = (int)(unsigned)(H * HEAD_DIM);                 // tensor_dim0_stride = H*D
      g1[6] = 0;
      g1[7] = 0;
      const i32x4 gz4 = {0, 0, 0, 0};
      const i32x8 gz8 = {0, 0, 0, 0, 0, 0, 0, 0};
      __builtin_amdgcn_tensor_load_to_lds(g0, g1, gz4, gz4, gz8, 0);
    }
#endif

    // ---- V tile: two adjacent keys packed per b32 store (cvt_pk + ds_store_b32) ----
    {
      const int kp2 = (lane) * 2;                 // even key 0..62
      const int db  = wave * 16;                  // 16-d slab per wave
      const float* vp0 = V + (((size_t)b * Sk + (k0 + kp2)) * H + h) * HEAD_DIM + db;
      const float* vp1 = vp0 + (size_t)H * HEAD_DIM;
      #pragma unroll
      for (int jj = 0; jj < 4; ++jj) {
        const float4 a = *(const float4*)(vp0 + 4 * jj);
        const float4 c = *(const float4*)(vp1 + 4 * jj);
        v2h p0; p0[0] = (_Float16)a.x; p0[1] = (_Float16)c.x;
        v2h p1; p1[0] = (_Float16)a.y; p1[1] = (_Float16)c.y;
        v2h p2; p2[0] = (_Float16)a.z; p2[1] = (_Float16)c.z;
        v2h p3; p3[0] = (_Float16)a.w; p3[1] = (_Float16)c.w;
        *(v2h*)&Vt[(db + 4 * jj + 0) * KTILE + kp2] = p0;
        *(v2h*)&Vt[(db + 4 * jj + 1) * KTILE + kp2] = p1;
        *(v2h*)&Vt[(db + 4 * jj + 2) * KTILE + kp2] = p2;
        *(v2h*)&Vt[(db + 4 * jj + 3) * KTILE + kp2] = p3;
      }
    }
    if (k0 + KTILE < kend) {
      __builtin_prefetch(K + (((size_t)b * Sk + (k0 + KTILE + (tid >> 2))) * H + h) * HEAD_DIM + (tid & 3) * 32, 0, 0);
      __builtin_prefetch(V + (((size_t)b * Sk + (k0 + KTILE + (tid >> 2))) * H + h) * HEAD_DIM + (tid & 3) * 32, 0, 0);
    }

#if HAVE_TDM
    if (wave == 0) __builtin_amdgcn_s_wait_tensorcnt((short)0);
    __syncthreads();
    // ---- vectorized LDS f32 -> LDS f16 conversion of the K tile ----
    {
      const int base = tid * 32;          // 256 threads x 32 f32 = 8192 elements
      #pragma unroll
      for (int i = 0; i < 4; ++i) {
        const float4 f0 = *(const float4*)(Kstage + base + 8 * i);
        const float4 f1 = *(const float4*)(Kstage + base + 8 * i + 4);
        *(v8h*)&Kf16[base + 8 * i] = cvt8(f0, f1);
      }
    }
#else
    // ---- fallback: direct global f32 -> f16 LDS K tile ----
    {
      const int kl = tid >> 2;
      const int dc = (tid & 3) * 32;
      const float* kp = K + (((size_t)b * Sk + (k0 + kl)) * H + h) * HEAD_DIM + dc;
      #pragma unroll
      for (int i = 0; i < 4; ++i) {
        const float4 f0 = *(const float4*)(kp + 8 * i);
        const float4 f1 = *(const float4*)(kp + 8 * i + 4);
        *(v8h*)&Kf16[kl * HEAD_DIM + dc + 8 * i] = cvt8(f0, f1);
      }
    }
#endif
    __syncthreads();

    // Waves whose 16 rows are entirely below this key block skip compute (not barriers).
    if (k0 <= q0 + 15) {
      // ---- S = Q K^T : 4 N-tiles x 4 K-chunks -> 16 WMMA ----
      v8f s[4];
      #pragma unroll
      for (int t = 0; t < 4; ++t) s[t] = zero;
      #pragma unroll
      for (int c = 0; c < 4; ++c) {
        const int koff = 32 * c + 8 * lhalf;
        #pragma unroll
        for (int t = 0; t < 4; ++t) {
          const _Float16* kr = &Kf16[(16 * t + l16) * HEAD_DIM + koff];
          const v16h bk = __builtin_shufflevector(*(const v8h*)kr, *(const v8h*)(kr + 16), 0,1,2,3,4,5,6,7,8,9,10,11,12,13,14,15);
          s[t] = __builtin_amdgcn_wmma_f32_16x16x32_f16(false, qfrag[c], false, bk, (short)0, s[t], false, false);
        }
      }

      // ---- ALiBi bias (incremental) + causal mask only on the diagonal block ----
      const float fn    = (float)l16;
      const float base0 = slope2 * ((float)(k0 - q0 - 8 * lhalf) + fn);
      const bool  needMask = (k0 + KTILE - 1 > q0);
      float lm[8];
      #pragma unroll
      for (int r = 0; r < 8; ++r) {
        const float br = base0 - (float)r * slope2;
        float e[4];
        e[0] = fmaf(s[0][r], scale2, br);
        e[1] = fmaf(s[1][r], scale2, br + slope16);
        e[2] = fmaf(s[2][r], scale2, br + 2.0f * slope16);
        e[3] = fmaf(s[3][r], scale2, br + 3.0f * slope16);
        if (needMask) {
          const int qi = q0 + r + 8 * lhalf;
          #pragma unroll
          for (int t = 0; t < 4; ++t)
            if (k0 + 16 * t + l16 > qi) e[t] = -1e30f;  // finite sentinel
        }
        #pragma unroll
        for (int t = 0; t < 4; ++t) s[t][r] = e[t];
        lm[r] = fmaxf(fmaxf(e[0], e[1]), fmaxf(e[2], e[3]));
      }

      // ---- row max: f16-packed butterflies, 2 rows per shuffle ----
      float rmax[8];
      #pragma unroll
      for (int p = 0; p < 4; ++p) {
        union { v2h h; int i; } u, v;
        u.h[0] = (_Float16)lm[2 * p];
        u.h[1] = (_Float16)lm[2 * p + 1];
        #pragma unroll
        for (int msk = 1; msk <= 8; msk <<= 1) {
          v.i = __shfl_xor(u.i, msk, 32);
          u.h = __builtin_elementwise_max(u.h, v.h);
        }
        rmax[2 * p]     = (float)u.h[0];
        rmax[2 * p + 1] = (float)u.h[1];
      }

      // ---- online update: rescale, exponentiate ----
      float alphas[8];
      #pragma unroll
      for (int r = 0; r < 8; ++r) {
        const float mnew = fmaxf(mrow[r], rmax[r]);
        alphas[r] = exp2f(mrow[r] - mnew);
        mrow[r]   = mnew;
        #pragma unroll
        for (int t = 0; t < 4; ++t) s[t][r] = exp2f(s[t][r] - mnew);
        #pragma unroll
        for (int t = 0; t < 8; ++t) o[t][r] *= alphas[r];
      }

      // ---- reshape P (C/D layout -> A layout) via per-wave LDS bounce ----
      _Float16* Pw = (_Float16*)(smem + PSH_OFF) + wave * (16 * KTILE);
      #pragma unroll
      for (int r = 0; r < 8; ++r) {
        const int m = r + 8 * lhalf;
        #pragma unroll
        for (int t = 0; t < 4; ++t)
          Pw[m * KTILE + 16 * t + l16] = (_Float16)s[t][r];
      }
      v16h pa[2];
      #pragma unroll
      for (int g = 0; g < 2; ++g) {
        const _Float16* pr = Pw + l16 * KTILE + 32 * g + 8 * lhalf;
        const v8h g0 = *(const v8h*)(pr);
        const v8h g1 = *(const v8h*)(pr + 16);
        pa[g] = __builtin_shufflevector(g0, g1, 0,1,2,3,4,5,6,7,8,9,10,11,12,13,14,15);
      }

      // ---- row sums via WMMA with all-ones B ----
      v8f ls = zero;
      ls = __builtin_amdgcn_wmma_f32_16x16x32_f16(false, pa[0], false, onesf, (short)0, ls, false, false);
      ls = __builtin_amdgcn_wmma_f32_16x16x32_f16(false, pa[1], false, onesf, (short)0, ls, false, false);
      #pragma unroll
      for (int r = 0; r < 8; ++r) lrow[r] = lrow[r] * alphas[r] + ls[r];

      // ---- O += P V : 8 d-tiles x 2 key-chunks -> 16 WMMA ----
      #pragma unroll
      for (int t = 0; t < 8; ++t) {
        #pragma unroll
        for (int g = 0; g < 2; ++g) {
          const _Float16* vr = &Vt[(16 * t + l16) * KTILE + 32 * g + 8 * lhalf];
          const v16h bv = __builtin_shufflevector(*(const v8h*)vr, *(const v8h*)(vr + 16), 0,1,2,3,4,5,6,7,8,9,10,11,12,13,14,15);
          o[t] = __builtin_amdgcn_wmma_f32_16x16x32_f16(false, pa[g], false, bv, (short)0, o[t], false, false);
        }
      }
    }
  }

  // ---- epilogue: normalize, store f32 BSHD ----
  #pragma unroll
  for (int r = 0; r < 8; ++r) {
    const int   qrow = q0 + r + 8 * lhalf;
    const float inv  = 1.0f / lrow[r];
    float* op = O + (((size_t)b * Sq + qrow) * H + h) * HEAD_DIM;
    #pragma unroll
    for (int t = 0; t < 8; ++t)
      op[16 * t + l16] = o[t][r] * inv;
  }
}

extern "C" void kernel_launch(void* const* d_in, const int* in_sizes, int n_in,
                              void* d_out, int out_size, void* d_ws, size_t ws_size,
                              hipStream_t stream) {
  const float* Q  = (const float*)d_in[0];
  const float* K  = (const float*)d_in[1];
  const float* V  = (const float*)d_in[2];
  const float* SL = (const float*)d_in[3];
  float* O = (float*)d_out;

  const int B  = 4;
  const int D  = HEAD_DIM;
  const int H  = in_sizes[3];
  const int Sq = in_sizes[0] / (B * H * D);
  const int Sk = in_sizes[1] / (B * H * D);

  dim3 grid(B * H * (Sq / QTILE));
  dim3 block(NTHREADS);
  fa_alibi_causal_kernel<<<grid, block, 0, stream>>>(Q, K, V, SL, O, B, Sq, Sk, H);
}
